// DNCCell_55293408968884
// MI455X (gfx1250) — matrix operand order repaired and checked by reading, not compiled, not run on hardware
//
#include <hip/hip_runtime.h>
#include <cstddef>
#include <cstdint>

#define Bsz 256
#define Tsz 256
#define Usz 512
#define Msz 64

typedef __attribute__((ext_vector_type(16))) __bf16 v16bf;
typedef __attribute__((ext_vector_type(8)))  float  v8f;

union BFrag { v16bf v; float4 f4[2]; };

// A-operand fragment (16x32 bf16) from row-major bf16 LDS tile, per ISA 7.12.2:
// lanes 0-15 hold rows M=0..15 with K = {0..7,16..23}; lanes 16-31 same rows, K = {8..15,24..31}
// Layout makes each half contiguous -> compiler emits 2x ds_load_b128.
__device__ __forceinline__ v16bf load_a_frag(const __bf16* S, int row, int hi, int k0) {
  v16bf a;
#pragma unroll
  for (int v = 0; v < 8; ++v) {
    int kb = k0 + ((v < 4) ? 0 : 16) + hi * 8 + 2 * (v & 3);
    a[2 * v]     = S[row * Usz + kb];
    a[2 * v + 1] = S[row * Usz + kb + 1];
  }
  return a;
}

// B-operand fragment, pre-packed so each lane reads its 16 bf16 contiguously (2x global b128)
__device__ __forceinline__ v16bf load_b_frag(const __bf16* __restrict__ Bfr, int nt, int ks, int lane) {
  const float4* p = reinterpret_cast<const float4*>(Bfr + (size_t)((nt * 16 + ks) * 32 + lane) * 16);
  BFrag u;
  u.f4[0] = p[0];
  u.f4[1] = p[1];
  return u.v;
}

// Software-pipelined single-N-tile GEMM: [16 x 512] (LDS bf16) @ [512 x 16] (packed B) -> v8f acc
__device__ __forceinline__ v8f gemm_tile(const __bf16* A, const __bf16* __restrict__ Bfr,
                                         int nt, int r16, int hi, int lane) {
  v8f acc = {};
  v16bf a = load_a_frag(A, r16, hi, 0);
  v16bf b = load_b_frag(Bfr, nt, 0, lane);
#pragma unroll
  for (int ks = 0; ks < 16; ++ks) {
    v16bf an, bn;
    if (ks < 15) {                       // prefetch next fragments before consuming current
      an = load_a_frag(A, r16, hi, (ks + 1) * 32);
      bn = load_b_frag(Bfr, nt, ks + 1, lane);
    }
    acc = __builtin_amdgcn_wmma_f32_16x16x32_bf16(false, a, false, b, (short)0, acc, false, false);
    if (ks < 15) { a = an; b = bn; }
  }
  return acc;
}

// Pack a row-major f32 [512 x ld] weight (cols 0..ntiles*16-1) into bf16 WMMA B-fragment order.
__global__ void pack_b_frags(const float* __restrict__ src, __bf16* __restrict__ dst, int ld, int total) {
  int idx = blockIdx.x * blockDim.x + threadIdx.x;
  if (idx >= total) return;
  int e    = idx & 15;
  int lane = (idx >> 4) & 31;
  int ks   = (idx >> 9) & 15;
  int nt   = idx >> 13;
  int hi   = lane >> 4;
  int k    = ks * 32 + ((e < 8) ? 0 : 8) + hi * 8 + e;
  int col  = nt * 16 + (lane & 15);
  dst[idx] = (__bf16)src[k * ld + col];
}

__global__ __launch_bounds__(512, 1)
void dnc_persistent(const float* __restrict__ x,
                    const float* __restrict__ Wr,   // f32 original, for column 64
                    const float* __restrict__ br,
                    const float* __restrict__ bt,
                    const float* __restrict__ bw,
                    const __bf16* __restrict__ WtB,
                    const __bf16* __restrict__ WrB,
                    const __bf16* __restrict__ WwB,
                    float* __restrict__ mem,
                    float* __restrict__ out) {
  __shared__ __bf16 s_mean[16 * Usz];   // 16 KB
  __shared__ __bf16 s_att [16 * Usz];   // 16 KB
  __shared__ __bf16 s_tr  [16 * Usz];   // 16 KB
  __shared__ float  s_log [16 * 68];    // logits -> softmax weights (65 used, padded)
  __shared__ float  s_ww  [16 * Msz];   // write gates

  const int blk  = blockIdx.x;          // batch tile: rows blk*16 .. blk*16+15
  const int tid  = threadIdx.x;         // 0..511 (u = tid in elementwise passes)
  const int wave = tid >> 5;            // 16 wave32s
  const int lane = tid & 31;
  const int r16  = lane & 15;
  const int hi   = lane >> 4;

  float* memb = mem + (size_t)blk * 16 * Msz * Usz;   // this tile's state, L2-resident
  const size_t xbase = (size_t)blk * 16 * Tsz * Usz;

  for (int t = 0; t < Tsz; ++t) {
    // ---- pass 1: mean over (64 memory rows + xt) -> bf16 LDS ----
    for (int b = 0; b < 16; ++b) {
      float s = x[xbase + ((size_t)b * Tsz + t) * Usz + tid];
      const float* mp = memb + (size_t)b * Msz * Usz + tid;
#pragma unroll 8
      for (int m = 0; m < Msz; ++m) s += mp[(size_t)m * Usz];
      s_mean[b * Usz + tid] = (__bf16)(s * (1.0f / 65.0f));
    }
    __syncthreads();

    // ---- pass 2: read logits = mean @ Wr + br (WMMA for cols 0..63, VALU for col 64) ----
    if (wave < 4) {
      v8f acc = gemm_tile(s_mean, WrB, wave, r16, hi, lane);
      const int col = wave * 16 + r16;
      const float bias = br[col];
#pragma unroll
      for (int r = 0; r < 8; ++r)
        s_log[(r + hi * 8) * 68 + col] = acc[r] + bias;
    } else if (wave == 4 && lane < 16) {
      float s = 0.0f;
      for (int u = 0; u < Usz; ++u)
        s += (float)s_mean[lane * Usz + u] * Wr[u * 65 + 64];
      s_log[lane * 68 + 64] = s + br[64];
    }
    __syncthreads();

    // ---- softmax over 65 (in place, one lane per batch row) ----
    if (wave == 0 && lane < 16) {
      float mx = -3.4e38f;
      for (int j = 0; j < 65; ++j) mx = fmaxf(mx, s_log[lane * 68 + j]);
      float sum = 0.0f;
      for (int j = 0; j < 65; ++j) {
        float e = __expf(s_log[lane * 68 + j] - mx);
        s_log[lane * 68 + j] = e;
        sum += e;
      }
      float inv = 1.0f / sum;
      for (int j = 0; j < 65; ++j) s_log[lane * 68 + j] *= inv;
    }
    __syncthreads();

    // ---- pass 3: attended = sum_m rw[m]*mem[m,:] + rw[64]*xt ----
    for (int b = 0; b < 16; ++b) {
      float s = s_log[b * 68 + 64] * x[xbase + ((size_t)b * Tsz + t) * Usz + tid];
      const float* mp = memb + (size_t)b * Msz * Usz + tid;
#pragma unroll 8
      for (int m = 0; m < Msz; ++m) s += s_log[b * 68 + m] * mp[(size_t)m * Usz];
      s_att[b * Usz + tid] = (__bf16)s;
    }
    __syncthreads();

    // ---- pass 4: transformed = relu(attended @ Wt + bt) ----
    // Each wave owns 2 N-tiles: one A-fragment feeds two back-to-back WMMAs (dual accumulators),
    // with next-iteration A/B fragments prefetched (software pipelining).
    {
      const int nt0 = wave * 2, nt1 = nt0 + 1;
      v8f acc0 = {}, acc1 = {};
      v16bf a  = load_a_frag(s_att, r16, hi, 0);
      v16bf b0 = load_b_frag(WtB, nt0, 0, lane);
      v16bf b1 = load_b_frag(WtB, nt1, 0, lane);
#pragma unroll
      for (int ks = 0; ks < 16; ++ks) {
        v16bf an, b0n, b1n;
        if (ks < 15) {
          an  = load_a_frag(s_att, r16, hi, (ks + 1) * 32);
          b0n = load_b_frag(WtB, nt0, ks + 1, lane);
          b1n = load_b_frag(WtB, nt1, ks + 1, lane);
        }
        acc0 = __builtin_amdgcn_wmma_f32_16x16x32_bf16(false, a, false, b0, (short)0, acc0, false, false);
        acc1 = __builtin_amdgcn_wmma_f32_16x16x32_bf16(false, a, false, b1, (short)0, acc1, false, false);
        if (ks < 15) { a = an; b0 = b0n; b1 = b1n; }
      }
      const int col0 = nt0 * 16 + r16;
      const int col1 = nt1 * 16 + r16;
      const float bias0 = bt[col0];
      const float bias1 = bt[col1];
#pragma unroll
      for (int r = 0; r < 8; ++r) {
        const int row = r + hi * 8;
        float v0 = fmaxf(acc0[r] + bias0, 0.0f);
        float v1 = fmaxf(acc1[r] + bias1, 0.0f);
        // streamed output: non-temporal so it doesn't evict the L2-resident state/weights
        __builtin_nontemporal_store(v0, &out[xbase + ((size_t)row * Tsz + t) * Usz + col0]);
        __builtin_nontemporal_store(v1, &out[xbase + ((size_t)row * Tsz + t) * Usz + col1]);
        s_tr[row * Usz + col0] = (__bf16)v0;
        s_tr[row * Usz + col1] = (__bf16)v1;
      }
    }
    __syncthreads();

    // ---- pass 5: write_w = sigmoid(transformed @ Ww + bw) ----
    if (wave < 4) {
      v8f acc = gemm_tile(s_tr, WwB, wave, r16, hi, lane);
      const int col = wave * 16 + r16;
      const float bias = bw[col];
#pragma unroll
      for (int r = 0; r < 8; ++r) {
        float v = acc[r] + bias;
        s_ww[(r + hi * 8) * Msz + col] = 1.0f / (1.0f + __expf(-v));
      }
    }
    __syncthreads();

    // ---- pass 6: mem = (1-w)*mem + w*transformed (linear, coalesced, stays in L2) ----
    for (int b = 0; b < 16; ++b) {
      const float tv = (float)s_tr[b * Usz + tid];
      float* mp = memb + (size_t)b * Msz * Usz + tid;
#pragma unroll 4
      for (int m = 0; m < Msz; ++m) {
        const float w  = s_ww[b * Msz + m];
        const float mv = mp[(size_t)m * Usz];
        mp[(size_t)m * Usz] = mv + w * (tv - mv);
      }
    }
    __syncthreads();
  }
}

extern "C" void kernel_launch(void* const* d_in, const int* in_sizes, int n_in,
                              void* d_out, int out_size, void* d_ws, size_t ws_size,
                              hipStream_t stream) {
  const float* x       = (const float*)d_in[0];   // [B,T,U]
  const float* memory0 = (const float*)d_in[1];   // [B, M*U]
  const float* Wr      = (const float*)d_in[2];   // [U, M+1]
  const float* br      = (const float*)d_in[3];   // [M+1]
  const float* Wt      = (const float*)d_in[4];   // [U, U]
  const float* bt      = (const float*)d_in[5];   // [U]
  const float* Ww      = (const float*)d_in[6];   // [U, M]
  const float* bw      = (const float*)d_in[7];   // [M]
  float* out = (float*)d_out;                     // [B,T,U]

  // workspace carve-up
  char* ws = (char*)d_ws;
  size_t off = 0;
  float*  mem = (float*)(ws + off);  off += (size_t)Bsz * Msz * Usz * sizeof(float);   // 33.5 MB, L2-resident
  __bf16* WtB = (__bf16*)(ws + off); off += (size_t)32 * 16 * 32 * 16 * sizeof(__bf16);
  __bf16* WrB = (__bf16*)(ws + off); off += (size_t)4  * 16 * 32 * 16 * sizeof(__bf16);
  __bf16* WwB = (__bf16*)(ws + off); off += (size_t)4  * 16 * 32 * 16 * sizeof(__bf16);

  // re-initialize state every call (deterministic), pack weights to bf16 fragments
  hipMemcpyAsync(mem, memory0, (size_t)Bsz * Msz * Usz * sizeof(float),
                 hipMemcpyDeviceToDevice, stream);
  pack_b_frags<<<(32 * 16 * 32 * 16 + 255) / 256, 256, 0, stream>>>(Wt, WtB, Usz, 32 * 16 * 32 * 16);
  pack_b_frags<<<(4  * 16 * 32 * 16 + 255) / 256, 256, 0, stream>>>(Wr, WrB, Msz + 1, 4 * 16 * 32 * 16);
  pack_b_frags<<<(4  * 16 * 32 * 16 + 255) / 256, 256, 0, stream>>>(Ww, WwB, Msz, 4 * 16 * 32 * 16);

  // persistent recurrence: one block per 16-row batch tile, whole T loop inside
  dnc_persistent<<<Bsz / 16, 512, 0, stream>>>(x, Wr, br, bt, bw, WtB, WrB, WwB, mem, out);
}
